// SeqAttention_54975581389362
// MI455X (gfx1250) — matrix operand order repaired and checked by reading, not compiled, not run on hardware
//
#include <hip/hip_runtime.h>

// Banded attention with positional bias, flash-style single pass.
// B=4 H=8 M=512 L=2048 D=64, KL=2560, out (32,512,64) f32.

#define BH_   32
#define MQ_   512
#define LW_   2048
#define DH_   64
#define KL_   2560

typedef __bf16 v16bf  __attribute__((ext_vector_type(16)));
typedef __bf16 bf16x8 __attribute__((ext_vector_type(8)));
typedef float  v8f    __attribute__((ext_vector_type(8)));
typedef float  f32x4  __attribute__((ext_vector_type(4)));

static __device__ inline v8f wmma_bf16(v16bf a, v16bf b, v8f c) {
  return __builtin_amdgcn_wmma_f32_16x16x32_bf16(false, a, false, b, (short)0, c, false, false);
}

// B operand (K=32 x N=16), 16-bit: lane(n=l&15, kh=l>>4) holds k = kh*8 + [0..7], 16 + kh*8 + [0..7],
// from LDS layout [n-row][k] row-major.
static __device__ inline v16bf ld_b_op(const __bf16* base, int pitch, int nbase, int kbase, int lane) {
  const int n = lane & 15, kh = lane >> 4;
  const __bf16* p = base + (size_t)(nbase + n) * pitch + kbase + kh * 8;
  union { v16bf v; bf16x8 h[2]; } u;
  u.h[0] = *(const bf16x8*)p;
  u.h[1] = *(const bf16x8*)(p + 16);
  return u.v;
}

// A operand (M=16 x K=32), 16-bit: lane(row=l&15, kh=l>>4):
//   elems 0..7  -> k = kh*8 + [0..7]         (VGPR0..3)
//   elems 8..15 -> k = 16 + kh*8 + [0..7]    (VGPR4..7)
static __device__ inline v16bf ld_a_op(const __bf16* base, int pitch, int kbase, int lane) {
  const int row = lane & 15, kh = lane >> 4;
  const __bf16* p = base + (size_t)row * pitch + kbase + kh * 8;
  union { v16bf v; bf16x8 h[2]; } u;
  u.h[0] = *(const bf16x8*)p;
  u.h[1] = *(const bf16x8*)(p + 16);
  return u.v;
}

// Q A-operand straight from global f32 row (one 16-row tile), converted to bf16.
static __device__ inline v16bf ld_q_op(const float* qrow, int dbase, int kh) {
  const float* p0 = qrow + dbase + kh * 8;
  const float* p1 = p0 + 16;
  f32x4 a0 = *(const f32x4*)(p0);
  f32x4 a1 = *(const f32x4*)(p0 + 4);
  f32x4 b0 = *(const f32x4*)(p1);
  f32x4 b1 = *(const f32x4*)(p1 + 4);
  v16bf r;
  r[0] = (__bf16)a0.x; r[1] = (__bf16)a0.y; r[2] = (__bf16)a0.z; r[3] = (__bf16)a0.w;
  r[4] = (__bf16)a1.x; r[5] = (__bf16)a1.y; r[6] = (__bf16)a1.z; r[7] = (__bf16)a1.w;
  r[8] = (__bf16)b0.x; r[9] = (__bf16)b0.y; r[10] = (__bf16)b0.z; r[11] = (__bf16)b0.w;
  r[12] = (__bf16)b1.x; r[13] = (__bf16)b1.y; r[14] = (__bf16)b1.z; r[15] = (__bf16)b1.w;
  return r;
}

__global__ __launch_bounds__(32)
void banded_attn_wmma_kernel(const float* __restrict__ query,
                             const float* __restrict__ key_,
                             const float* __restrict__ value,
                             const float* __restrict__ key_pe,
                             const float* __restrict__ current_val,
                             float* __restrict__ out) {
  // LDS (pitches padded against bank conflicts); ~45 KB total
  __shared__ __align__(16) __bf16 Kl[80 * 72];    // K slab  [k-row][d]
  __shared__ __align__(16) __bf16 Vt[64 * 104];   // V slab  [d][k]   (k 80..95 stay zero)
  __shared__ __align__(16) __bf16 PEl[64 * 72];   // PE slab [j][d]
  __shared__ __align__(16) float  Pbuf[16 * 64];  // Q@PE    [m][jloc]
  __shared__ __align__(16) float  Sbuf[16 * 68];  // scores  [m][jloc]
  __shared__ __align__(16) __bf16 Abuf[16 * 104]; // probs   [m][n-m0-j0] skewed (80..103 zero)

  const int lane = threadIdx.x;
  const int kh = lane >> 4, nl = lane & 15;
  const int bh = blockIdx.x >> 5;        // 32 m-tiles per bh
  const int m0 = (blockIdx.x & 31) << 4;

  // one-time zero of the permanently-padded regions
  for (int i = lane; i < 16 * 104; i += 32) Abuf[i] = (__bf16)0.0f;
  for (int i = lane; i < 64 * 104; i += 32) Vt[i] = (__bf16)0.0f;

  const float* qrow = query + ((size_t)bh * MQ_ + m0 + nl) * DH_;
  v16bf q0 = ld_q_op(qrow, 0, kh);
  v16bf q1 = ld_q_op(qrow, 32, kh);

  v8f O0 = {}, O1 = {}, O2 = {}, O3 = {};
  float mrun = -1e30f, Zrun = 0.0f, Zmrun = 0.0f;

  const float cv = current_val[bh & 7];         // h = bh % H
  const float mc = cv * 2048.0f - 1983.0f;      // mask = clamp((j+mc)/64, 0, 1)

  __syncthreads();

  for (int j0 = 0; j0 < LW_; j0 += 64) {
    const int nbase = m0 + j0;

    // ---- stage K slab: rows nbase..nbase+79, f32 -> bf16 row-major ----
    #pragma unroll 4
    for (int it = 0; it < 40; ++it) {
      int idx = it * 32 + lane;
      int row = idx >> 4, d4 = (idx & 15) << 2;
      f32x4 x = *(const f32x4*)(key_ + ((size_t)bh * KL_ + nbase + row) * DH_ + d4);
      __bf16* dst = &Kl[row * 72 + d4];
      dst[0] = (__bf16)x.x; dst[1] = (__bf16)x.y; dst[2] = (__bf16)x.z; dst[3] = (__bf16)x.w;
    }
    // ---- stage V slab transposed: Vt[d][k] ----
    #pragma unroll 4
    for (int it = 0; it < 40; ++it) {
      int idx = it * 32 + lane;
      int row = idx >> 4, d4 = (idx & 15) << 2;
      f32x4 x = *(const f32x4*)(value + ((size_t)bh * KL_ + nbase + row) * DH_ + d4);
      Vt[(d4 + 0) * 104 + row] = (__bf16)x.x;
      Vt[(d4 + 1) * 104 + row] = (__bf16)x.y;
      Vt[(d4 + 2) * 104 + row] = (__bf16)x.z;
      Vt[(d4 + 3) * 104 + row] = (__bf16)x.w;
    }
    // ---- stage PE chunk transposed: PEl[j][d]  (key_pe is [d][L]) ----
    #pragma unroll 4
    for (int it = 0; it < 32; ++it) {
      int idx = it * 32 + lane;
      int d = idx >> 4, j4 = (idx & 15) << 2;
      f32x4 x = *(const f32x4*)(key_pe + (size_t)d * LW_ + j0 + j4);
      PEl[(j4 + 0) * 72 + d] = (__bf16)x.x;
      PEl[(j4 + 1) * 72 + d] = (__bf16)x.y;
      PEl[(j4 + 2) * 72 + d] = (__bf16)x.z;
      PEl[(j4 + 3) * 72 + d] = (__bf16)x.w;
    }
    __syncthreads();

    // ---- positional term: P = Q @ PE chunk (j-space), f32 into Pbuf ----
    #pragma unroll
    for (int jt = 0; jt < 4; ++jt) {
      v8f P = {};
      P = wmma_bf16(q0, ld_b_op(PEl, 72, jt * 16, 0, lane), P);
      P = wmma_bf16(q1, ld_b_op(PEl, 72, jt * 16, 32, lane), P);
      #pragma unroll
      for (int r = 0; r < 8; ++r)
        Pbuf[(r + 8 * kh) * 64 + jt * 16 + nl] = P[r];
    }
    __syncthreads();

    // ---- content term in n-space (5 tiles cover the 80-wide slab);
    //      add skewed PE, scale, band-window -> Sbuf[m][jloc] ----
    #pragma unroll
    for (int t = 0; t < 5; ++t) {
      v8f S = {};
      S = wmma_bf16(q0, ld_b_op(Kl, 72, t * 16, 0, lane), S);
      S = wmma_bf16(q1, ld_b_op(Kl, 72, t * 16, 32, lane), S);
      #pragma unroll
      for (int r = 0; r < 8; ++r) {
        int m = r + 8 * kh;            // local query row
        int c = t * 16 + nl - m;       // j - j0
        if (c >= 0 && c < 64)
          Sbuf[m * 68 + c] = (S[r] + Pbuf[m * 64 + c]) * 0.125f;
      }
    }
    __syncthreads();

    // ---- flash softmax over the chunk; lane pair (nl, nl+16) owns row nl ----
    float sv[32];
    float tmax = -1e30f;
    const float* srow = &Sbuf[nl * 68 + kh * 32];
    #pragma unroll
    for (int i = 0; i < 32; ++i) { sv[i] = srow[i]; tmax = fmaxf(tmax, sv[i]); }
    tmax = fmaxf(tmax, __shfl_xor(tmax, 16, 32));
    const float mnew = fmaxf(mrun, tmax);
    const float alpha = __expf(mrun - mnew);
    float zs = 0.0f, zms = 0.0f;
    __bf16* arow = &Abuf[nl * 104 + nl + kh * 32];   // skew: col = (m-m0) + jloc
    #pragma unroll
    for (int i = 0; i < 32; ++i) {
      float p = __expf(sv[i] - mnew);
      float j = (float)(j0 + kh * 32 + i);
      float msk = fminf(fmaxf((j + mc) * 0.015625f, 0.0f), 1.0f);
      float pm = p * msk;
      zs += p; zms += pm;
      arow[i] = (__bf16)pm;
    }
    // zero the skew edges of this row for this chunk
    if (kh == 0) { for (int c2 = 0; c2 < nl; ++c2)        Abuf[nl * 104 + c2] = (__bf16)0.0f; }
    else         { for (int c2 = nl + 64; c2 < 80; ++c2)  Abuf[nl * 104 + c2] = (__bf16)0.0f; }
    zs  += __shfl_xor(zs, 16, 32);
    zms += __shfl_xor(zms, 16, 32);
    Zrun  = Zrun * alpha + zs;
    Zmrun = Zmrun * alpha + zms;
    mrun = mnew;
    __syncthreads();

    // ---- rescale running O by per-row alpha (bpermute into C layout) ----
    #pragma unroll
    for (int r = 0; r < 8; ++r) {
      float a = __shfl(alpha, r + 8 * kh, 32);
      O0[r] *= a; O1[r] *= a; O2[r] *= a; O3[r] *= a;
    }
    // ---- O += A_skew(16x96) @ V(96x64): 3 K-steps x 4 d-tiles ----
    #pragma unroll
    for (int ks = 0; ks < 3; ++ks) {
      v16bf a = ld_a_op(Abuf, 104, ks * 32, lane);
      O0 = wmma_bf16(a, ld_b_op(Vt, 104,  0, ks * 32, lane), O0);
      O1 = wmma_bf16(a, ld_b_op(Vt, 104, 16, ks * 32, lane), O1);
      O2 = wmma_bf16(a, ld_b_op(Vt, 104, 32, ks * 32, lane), O2);
      O3 = wmma_bf16(a, ld_b_op(Vt, 104, 48, ks * 32, lane), O3);
    }
    __syncthreads();
  }

  // out = O / (Zm + 1e-8 * Z)   (exact mask+renorm semantics)
  const float inv = 1.0f / (Zmrun + 1e-8f * Zrun);
  #pragma unroll
  for (int r = 0; r < 8; ++r) {
    float w = __shfl(inv, r + 8 * kh, 32);
    float* orow = out + ((size_t)bh * MQ_ + m0 + r + 8 * kh) * DH_ + nl;
    orow[0]  = O0[r] * w;
    orow[16] = O1[r] * w;
    orow[32] = O2[r] * w;
    orow[48] = O3[r] * w;
  }
}

extern "C" void kernel_launch(void* const* d_in, const int* in_sizes, int n_in,
                              void* d_out, int out_size, void* d_ws, size_t ws_size,
                              hipStream_t stream) {
  (void)in_sizes; (void)n_in; (void)out_size; (void)d_ws; (void)ws_size;
  const float* query       = (const float*)d_in[0];
  const float* key_        = (const float*)d_in[1];
  const float* value       = (const float*)d_in[2];
  const float* key_pe      = (const float*)d_in[3];
  const float* current_val = (const float*)d_in[4];
  float* out = (float*)d_out;

  dim3 grid(BH_ * (MQ_ / 16));  // 1024 tiles, one wave32 each
  dim3 block(32);
  banded_attn_wmma_kernel<<<grid, block, 0, stream>>>(query, key_, value, key_pe,
                                                      current_val, out);
}
